// CustomTransformerEncoderLayer_47141561041116
// MI455X (gfx1250) — matrix operand order repaired and checked by reading, not compiled, not run on hardware
//
#include <hip/hip_runtime.h>
#include <hip/hip_bf16.h>

// ---------------------------------------------------------------------------
// Types for CDNA5 WMMA (wave32): V_WMMA_F32_16X16X32_F16
// ---------------------------------------------------------------------------
typedef _Float16 h8  __attribute__((ext_vector_type(8)));
typedef _Float16 h16 __attribute__((ext_vector_type(16)));
typedef float    f8  __attribute__((ext_vector_type(8)));
typedef int      i4  __attribute__((ext_vector_type(4)));

__device__ __forceinline__ h16 join8(h8 a, h8 b) {
  h16 r;
#pragma unroll
  for (int i = 0; i < 8; ++i) { r[i] = a[i]; r[i + 8] = b[i]; }
  return r;
}

__device__ __forceinline__ f8 zf8() {
  f8 r;
#pragma unroll
  for (int i = 0; i < 8; ++i) r[i] = 0.0f;
  return r;
}

__device__ __forceinline__ f8 wmma_f16(h16 a, h16 b, f8 c) {
  // 8 args: (neg_a, A, neg_b, B, c_mod, C, reuse_a, reuse_b)
  return __builtin_amdgcn_wmma_f32_16x16x32_f16(false, a, false, b, (short)0, c,
                                                false, false);
}

// ---------------------------------------------------------------------------
// Async global -> LDS (CDNA5 GLOBAL_LOAD_ASYNC_TO_LDS_B128, ASYNCcnt-tracked).
// Probe confirmed: 4 args, param0 = int4 addrspace(1)*, so pass int4 pointers.
// ---------------------------------------------------------------------------
#if defined(__gfx1250__) && __has_builtin(__builtin_amdgcn_global_load_async_to_lds_b128) && __has_builtin(__builtin_amdgcn_s_wait_asynccnt)
#define HAVE_ASYNC 1
#else
#define HAVE_ASYNC 0
#endif

__device__ __forceinline__ void g2l_b128(const _Float16* g, _Float16* l) {
#if HAVE_ASYNC
  __builtin_amdgcn_global_load_async_to_lds_b128(
      (__attribute__((address_space(1))) i4*)(void*)g,
      (__attribute__((address_space(3))) i4*)(void*)l, 0, 0);
#else
  *(h8*)l = *(const h8*)g;
#endif
}

__device__ __forceinline__ void async_wait() {
#if HAVE_ASYNC
  __builtin_amdgcn_s_wait_asynccnt(0);
#endif
}

// ---------------------------------------------------------------------------
// Problem constants (reference is fixed-shape)
// ---------------------------------------------------------------------------
#define CB 4
#define CS 2048
#define CD 512
#define CH 8
#define CHD 64
#define CF 2048
#define CM (CB * CS)   // 8192 rows

// ---------------------------------------------------------------------------
// f32 -> f16 conversion
// ---------------------------------------------------------------------------
__global__ void __launch_bounds__(256) f32_to_f16_k(const float* __restrict__ in,
                                                    _Float16* __restrict__ out,
                                                    int n) {
  int i = blockIdx.x * 256 + threadIdx.x;
  if (i < n) out[i] = (_Float16)in[i];
}

// ---------------------------------------------------------------------------
// Generic WMMA GEMM: C[M,N] = A[M,K] (f16) * B[K,N] (f16) + epilogue
// Block = 256 thr (8 waves), tile 128x128, BK=32, double-buffered LDS.
// Wave w: sub-tile (wm = (w&3)*32, wn = (w>>2)*64) -> 2x4 fragments of 16x16.
// EPI 0: out f32 = acc + bias[col] + res[row*N+col]
// EPI 1: out f16 = relu(acc + bias[col])
// EPI 2: QKV scatter -> q/k/v f16 buffers laid out [B,H,S,HD]
// ---------------------------------------------------------------------------
template <int EPI>
__global__ void __launch_bounds__(256)
gemm_wmma(const _Float16* __restrict__ A, const _Float16* __restrict__ Bm,
          const float* __restrict__ bias, const float* __restrict__ res,
          float* __restrict__ outF, _Float16* __restrict__ outH,
          _Float16* __restrict__ qb, _Float16* __restrict__ kb,
          _Float16* __restrict__ vb, int M, int N, int K) {
  __shared__ __align__(16) _Float16 at[2][128 * 32];  // A tile [m][k]
  __shared__ __align__(16) _Float16 bt[2][128 * 32];  // B tile transposed [n][k]

  const int t = threadIdx.x;
  const int lane = t & 31, w = t >> 5;
  const int lh = lane >> 4, ll = lane & 15;       // lane half / lane-in-half
  const int wm = (w & 3) * 32, wn = (w >> 2) * 64;
  const int rowBase = blockIdx.x * 128, colBase = blockIdx.y * 128;

  f8 acc[2][4];
#pragma unroll
  for (int mi = 0; mi < 2; ++mi)
#pragma unroll
    for (int ni = 0; ni < 4; ++ni) acc[mi][ni] = zf8();

  const int arow = t >> 1, aoff = (t & 1) * 16;   // A staging: 16 halves/thread
  const int bk = t >> 3, bn0 = (t & 7) * 16;      // B staging: 16 halves/thread

  auto stage = [&](int buf, int k0) {
    // A tile: straight copy (async path)
    const _Float16* ga = A + (size_t)(rowBase + arow) * K + k0 + aoff;
    g2l_b128(ga,     &at[buf][arow * 32 + aoff]);
    g2l_b128(ga + 8, &at[buf][arow * 32 + aoff + 8]);
    // B tile transposed ([n][k] so fragment loads are contiguous)
    const h8* g = (const h8*)(Bm + (size_t)(k0 + bk) * N + colBase + bn0);
    h8 b0 = g[0], b1 = g[1];
#pragma unroll
    for (int j = 0; j < 8; ++j) {
      bt[buf][(bn0 + j) * 32 + bk]     = b0[j];
      bt[buf][(bn0 + 8 + j) * 32 + bk] = b1[j];
    }
  };

  const int nK = K >> 5;
  stage(0, 0);
  for (int it = 0; it < nK; ++it) {
    async_wait();           // async-LDS ops are not covered by dscnt waits
    __syncthreads();
    if (it + 1 < nK) stage((it + 1) & 1, (it + 1) << 5);
    const _Float16* atc = at[it & 1];
    const _Float16* btc = bt[it & 1];

    // A fragments: lane ll = row; elems 0-7: K = lh*8, elems 8-15: +16
    h16 af[2];
#pragma unroll
    for (int mi = 0; mi < 2; ++mi) {
      const h8* p = (const h8*)&atc[(wm + mi * 16 + ll) * 32 + lh * 8];
      af[mi] = join8(p[0], p[2]);  // p[2] == +16 halves
    }
    // B fragments: lane ll = col; elems 0-7: K = lh*8, elems 8-15: +16
#pragma unroll
    for (int ni = 0; ni < 4; ++ni) {
      const h8* p = (const h8*)&btc[(wn + ni * 16 + ll) * 32 + lh * 8];
      h16 bf = join8(p[0], p[2]);
#pragma unroll
      for (int mi = 0; mi < 2; ++mi)
        acc[mi][ni] = wmma_f16(af[mi], bf, acc[mi][ni]);
    }
  }

  // ---- epilogue; C layout: elem i, lane l -> row = i + 8*lh, col = ll ----
#pragma unroll
  for (int mi = 0; mi < 2; ++mi) {
#pragma unroll
    for (int ni = 0; ni < 4; ++ni) {
      const int c = colBase + wn + ni * 16 + ll;
      const float bia = bias[c];
#pragma unroll
      for (int i = 0; i < 8; ++i) {
        const int r = rowBase + wm + mi * 16 + i + lh * 8;
        float v = acc[mi][ni][i] + bia;
        if (EPI == 0) {
          outF[(size_t)r * N + c] = v + res[(size_t)r * N + c];
        } else if (EPI == 1) {
          outH[(size_t)r * N + c] = (_Float16)(v > 0.0f ? v : 0.0f);
        } else {
          const int part = c >> 9, hh = (c >> 6) & 7, hd = c & 63;
          const int bb = r >> 11, s0 = r & 2047;
          const size_t idx = (((size_t)(bb * CH + hh)) * CS + s0) * CHD + hd;
          _Float16* dst = (part == 0) ? qb : (part == 1) ? kb : vb;
          dst[idx] = (_Float16)v;
        }
      }
    }
  }
}

// ---------------------------------------------------------------------------
// Flash attention (non-causal, online softmax), f16 in / f16 out, f32 state.
// Grid: (B*H, S/64). Block: 128 thr = 4 waves, wave = 16 queries.
// 128-key chunks: 16 WMMA for Q*K^T + 16 WMMA for P*V per chunk; softmax
// shuffle reductions amortized over 128 keys (4x fewer ds_bpermute rounds).
// ---------------------------------------------------------------------------
__global__ void __launch_bounds__(128)
flash_attn(const _Float16* __restrict__ Q, const _Float16* __restrict__ Kb,
           const _Float16* __restrict__ Vb, _Float16* __restrict__ ctx) {
  __shared__ __align__(16) _Float16 kt[128 * 64];      // K chunk [key][hd]   16KB
  __shared__ __align__(16) _Float16 vt[64 * 128];      // V chunk T [hd][key] 16KB
  __shared__ __align__(16) _Float16 pt[4][16 * 128];   // per-wave P [q][key] 16KB

  const int t = threadIdx.x, lane = t & 31, w = t >> 5;
  const int lh = lane >> 4, ll = lane & 15;
  const int bh = blockIdx.x;                 // 0..31  (b*H + h)
  const int q0 = blockIdx.y * 64;            // query block base
  const int bb = bh >> 3, hh = bh & 7;
  const size_t base = (size_t)bh * CS * CHD;
  const float scale = 0.125f;                // 1/sqrt(64)

  // Q fragments (A layout) straight from global f16
  h16 qf[2];
  const int qrow = q0 + w * 16 + ll;
#pragma unroll
  for (int ka = 0; ka < 2; ++ka) {
    const h8* p = (const h8*)(Q + base + (size_t)qrow * CHD + ka * 32 + lh * 8);
    qf[ka] = join8(p[0], p[2]);
  }

  f8 O[4];
#pragma unroll
  for (int ni = 0; ni < 4; ++ni) O[ni] = zf8();
  float mrun[8], lsum[8];
#pragma unroll
  for (int i = 0; i < 8; ++i) { mrun[i] = -3.0e38f; lsum[i] = 0.0f; }

  for (int j0 = 0; j0 < CS; j0 += 128) {
    async_wait();
    __syncthreads();
    {  // stage K chunk [128][64]: thread t copies key row (j0+t), async path
      const _Float16* g = Kb + base + (size_t)(j0 + t) * CHD;
#pragma unroll
      for (int c = 0; c < 8; ++c) g2l_b128(g + c * 8, &kt[t * 64 + c * 8]);
    }
    {  // stage V chunk transposed -> vt[hd*128 + key]
      const h8* g = (const h8*)(Vb + base + (size_t)(j0 + t) * CHD);
#pragma unroll
      for (int c = 0; c < 8; ++c) {
        h8 a = g[c];
#pragma unroll
        for (int j = 0; j < 8; ++j) vt[(c * 8 + j) * 128 + t] = a[j];
      }
    }
    async_wait();
    __syncthreads();

    // ---- scores: S[16q x 128k] as eight 16x16 C fragments ----
    f8 s[8];
#pragma unroll
    for (int kc = 0; kc < 8; ++kc) {
      f8 a = zf8();
#pragma unroll
      for (int ka = 0; ka < 2; ++ka) {
        const h8* p = (const h8*)&kt[(kc * 16 + ll) * 64 + ka * 32 + lh * 8];
        h16 bf = join8(p[0], p[2]);
        a = wmma_f16(qf[ka], bf, a);
      }
      s[kc] = a;
    }

    // ---- online softmax: rows live per (elem i, lane-half) -> shfl_xor<16 ----
#pragma unroll
    for (int kc = 0; kc < 8; ++kc)
#pragma unroll
      for (int i = 0; i < 8; ++i) s[kc][i] *= scale;

    float fac[8];
#pragma unroll
    for (int i = 0; i < 8; ++i) {
      float cm = s[0][i];
#pragma unroll
      for (int kc = 1; kc < 8; ++kc) cm = fmaxf(cm, s[kc][i]);
      cm = fmaxf(cm, __shfl_xor(cm, 1));
      cm = fmaxf(cm, __shfl_xor(cm, 2));
      cm = fmaxf(cm, __shfl_xor(cm, 4));
      cm = fmaxf(cm, __shfl_xor(cm, 8));
      const float nm = fmaxf(mrun[i], cm);
      fac[i] = __expf(mrun[i] - nm);
      mrun[i] = nm;
      float rs = 0.0f;
#pragma unroll
      for (int kc = 0; kc < 8; ++kc) {
        const float e = __expf(s[kc][i] - nm);
        s[kc][i] = e;
        rs += e;
      }
      rs += __shfl_xor(rs, 1);
      rs += __shfl_xor(rs, 2);
      rs += __shfl_xor(rs, 4);
      rs += __shfl_xor(rs, 8);
      lsum[i] = lsum[i] * fac[i] + rs;
      O[0][i] *= fac[i]; O[1][i] *= fac[i]; O[2][i] *= fac[i]; O[3][i] *= fac[i];
    }

    // ---- C-layout -> A-layout via per-wave LDS round trip ----
#pragma unroll
    for (int kc = 0; kc < 8; ++kc)
#pragma unroll
      for (int i = 0; i < 8; ++i)
        pt[w][(i + lh * 8) * 128 + kc * 16 + ll] = (_Float16)s[kc][i];
    h16 pf[4];
#pragma unroll
    for (int kp = 0; kp < 4; ++kp) {
      const h8* p = (const h8*)&pt[w][ll * 128 + kp * 32 + lh * 8];
      pf[kp] = join8(p[0], p[2]);
    }

    // ---- O += P * V ----
#pragma unroll
    for (int ni = 0; ni < 4; ++ni) {
#pragma unroll
      for (int kp = 0; kp < 4; ++kp) {
        const h8* p = (const h8*)&vt[(ni * 16 + ll) * 128 + kp * 32 + lh * 8];
        h16 vf = join8(p[0], p[2]);
        O[ni] = wmma_f16(pf[kp], vf, O[ni]);
      }
    }
  }

  // ---- normalize and write ctx f16 [B*S, D] (col = h*64 + hd) ----
#pragma unroll
  for (int ni = 0; ni < 4; ++ni)
#pragma unroll
    for (int i = 0; i < 8; ++i) {
      const int q = q0 + w * 16 + i + lh * 8;
      const size_t r = (size_t)bb * CS + q;
      const int col = hh * CHD + ni * 16 + ll;
      ctx[r * CD + col] = (_Float16)(O[ni][i] / lsum[i]);
    }
}

// ---------------------------------------------------------------------------
// LayerNorm over rows of 512. Block = 256 thr (2 cols/thread), grid = rows.
// WH: also emit f16 copy for the next WMMA GEMM's A matrix.
// ---------------------------------------------------------------------------
template <bool WH>
__global__ void __launch_bounds__(256)
layernorm_k(const float* __restrict__ in, const float* __restrict__ g,
            const float* __restrict__ be, float* __restrict__ out,
            _Float16* __restrict__ outh) {
  __shared__ float red[16];
  const int row = blockIdx.x, t = threadIdx.x;
  const float* x = in + (size_t)row * CD;
  const float v0 = x[t], v1 = x[t + 256];
  float s = v0 + v1, ss = v0 * v0 + v1 * v1;
#pragma unroll
  for (int m = 1; m < 32; m <<= 1) {
    s += __shfl_xor(s, m);
    ss += __shfl_xor(ss, m);
  }
  if ((t & 31) == 0) { red[t >> 5] = s; red[8 + (t >> 5)] = ss; }
  __syncthreads();
  float ts = 0.0f, tss = 0.0f;
#pragma unroll
  for (int i = 0; i < 8; ++i) { ts += red[i]; tss += red[8 + i]; }
  const float mu = ts * (1.0f / CD);
  const float var = tss * (1.0f / CD) - mu * mu;
  const float rstd = rsqrtf(var + 1.0e-5f);
  const float o0 = (v0 - mu) * rstd * g[t] + be[t];
  const float o1 = (v1 - mu) * rstd * g[t + 256] + be[t + 256];
  float* o = out + (size_t)row * CD;
  o[t] = o0;
  o[t + 256] = o1;
  if (WH) {
    _Float16* oh = outh + (size_t)row * CD;
    oh[t] = (_Float16)o0;
    oh[t + 256] = (_Float16)o1;
  }
}

// ---------------------------------------------------------------------------
// Host-side orchestration
// ---------------------------------------------------------------------------
static inline void cvt(const float* a, _Float16* b, int n, hipStream_t st) {
  f32_to_f16_k<<<(n + 255) / 256, 256, 0, st>>>(a, b, n);
}

extern "C" void kernel_launch(void* const* d_in, const int* in_sizes, int n_in,
                              void* d_out, int out_size, void* d_ws,
                              size_t ws_size, hipStream_t stream) {
  (void)in_sizes; (void)n_in; (void)out_size; (void)ws_size;
  const float* x     = (const float*)d_in[0];
  const float* Wqkv  = (const float*)d_in[1];
  const float* bqkv  = (const float*)d_in[2];
  const float* Wo    = (const float*)d_in[3];
  const float* bo    = (const float*)d_in[4];
  const float* g1    = (const float*)d_in[5];
  const float* beta1 = (const float*)d_in[6];
  const float* W1    = (const float*)d_in[7];
  const float* b1    = (const float*)d_in[8];
  const float* W2    = (const float*)d_in[9];
  const float* b2    = (const float*)d_in[10];
  const float* g2    = (const float*)d_in[11];
  const float* beta2 = (const float*)d_in[12];

  char* ws = (char*)d_ws;
  size_t off = 0;
  auto alloc = [&](size_t bytes) -> void* {
    void* p = ws + off;
    off = (off + bytes + 255) & ~(size_t)255;
    return p;
  };
  _Float16* xh    = (_Float16*)alloc((size_t)CM * CD * 2);
  _Float16* Wqkvh = (_Float16*)alloc((size_t)CD * 3 * CD * 2);
  _Float16* Woh   = (_Float16*)alloc((size_t)CD * CD * 2);
  _Float16* W1h   = (_Float16*)alloc((size_t)CD * CF * 2);
  _Float16* W2h   = (_Float16*)alloc((size_t)CF * CD * 2);
  _Float16* qb    = (_Float16*)alloc((size_t)CM * CD * 2);
  _Float16* kb    = (_Float16*)alloc((size_t)CM * CD * 2);
  _Float16* vb    = (_Float16*)alloc((size_t)CM * CD * 2);
  _Float16* ctxh  = (_Float16*)alloc((size_t)CM * CD * 2);
  float*    y1    = (float*)alloc((size_t)CM * CD * 4);
  float*    x1f   = (float*)alloc((size_t)CM * CD * 4);
  _Float16* x1h   = (_Float16*)alloc((size_t)CM * CD * 2);
  _Float16* hid   = (_Float16*)alloc((size_t)CM * CF * 2);
  float*    y2    = (float*)alloc((size_t)CM * CD * 4);

  // 1) f32 -> f16 staging
  cvt(x, xh, CM * CD, stream);
  cvt(Wqkv, Wqkvh, CD * 3 * CD, stream);
  cvt(Wo, Woh, CD * CD, stream);
  cvt(W1, W1h, CD * CF, stream);
  cvt(W2, W2h, CF * CD, stream);

  // 2) QKV projection with scatter to [B,H,S,HD] f16 buffers
  gemm_wmma<2><<<dim3(CM / 128, (3 * CD) / 128), 256, 0, stream>>>(
      xh, Wqkvh, bqkv, nullptr, nullptr, nullptr, qb, kb, vb, CM, 3 * CD, CD);

  // 3) Flash attention -> ctx f16 [B*S, D]
  flash_attn<<<dim3(CB * CH, CS / 64), 128, 0, stream>>>(qb, kb, vb, ctxh);

  // 4) Output projection + residual (f32)
  gemm_wmma<0><<<dim3(CM / 128, CD / 128), 256, 0, stream>>>(
      ctxh, Woh, bo, x, y1, nullptr, nullptr, nullptr, nullptr, CM, CD, CD);

  // 5) LayerNorm 1 (f32 + f16 copies)
  layernorm_k<true><<<CM, 256, 0, stream>>>(y1, g1, beta1, x1f, x1h);

  // 6) FFN up + ReLU (f16 out)
  gemm_wmma<1><<<dim3(CM / 128, CF / 128), 256, 0, stream>>>(
      x1h, W1h, b1, nullptr, nullptr, hid, nullptr, nullptr, nullptr, CM, CF, CD);

  // 7) FFN down + residual (f32)
  gemm_wmma<0><<<dim3(CM / 128, CD / 128), 256, 0, stream>>>(
      hid, W2h, b2, x1f, y2, nullptr, nullptr, nullptr, nullptr, CM, CD, CF);

  // 8) LayerNorm 2 -> output
  layernorm_k<false><<<CM, 256, 0, stream>>>(y2, g2, beta2, (float*)d_out,
                                             nullptr);
}
